// Mask2formerLoss_8272107012749
// MI455X (gfx1250) — matrix operand order repaired and checked by reading, not compiled, not run on hardware
//
#include <hip/hip_runtime.h>
#include <hip/hip_bf16.h>
#include <stdint.h>

// ---------------- problem constants (from reference) ----------------
#define Bc     2
#define Qc     100
#define Tc     10
#define Hc     64
#define UHc    256
#define Pc     12544      // NUM_POINTS
#define NSc    37632      // N_SAMPLED = 3 * P
#define KIMP   9408       // 0.75 * P
#define KRANDc 3136       // P - KIMP
#define Nc     (Bc*Tc)    // 20 matched pairs
#define QT     7          // ceil(Q/16) query tiles
#define LAYERS 10

typedef __attribute__((ext_vector_type(16))) _Float16 v16h;
typedef __attribute__((ext_vector_type(8)))  float    v8f;

// ---------------- deterministic counter-based RNG ----------------
__device__ __forceinline__ uint32_t hash32(uint32_t x) {
    x ^= x >> 16; x *= 0x7feb352dU;
    x ^= x >> 15; x *= 0x846ca68bU;
    x ^= x >> 16; return x;
}
__device__ __forceinline__ float rng_u01(uint32_t tag, uint32_t idx, uint32_t d) {
    uint32_t h = hash32(tag ^ 0x9E3779B9U);
    h = hash32(h ^ idx);
    h = hash32(h ^ (d * 0x85EBCA6BU + 0x165667B1U));
    return (h >> 8) * (1.0f / 16777216.0f);
}

// ---------------- math helpers (fast variants for hot paths) ----------------
__device__ __forceinline__ float softplusf(float z) {
    return fmaxf(z, 0.0f) + log1pf(__expf(-fabsf(z)));
}
__device__ __forceinline__ float sigmoidf(float z) {
    return __fdividef(1.0f, 1.0f + __expf(-z));
}

// bilinear on 64x64 with edge clamp (jax.image.resize half-pixel convention)
__device__ __forceinline__ float sample64_clamp(const float* img, float sx, float sy) {
    sx = fminf(fmaxf(sx, 0.0f), 63.0f);
    sy = fminf(fmaxf(sy, 0.0f), 63.0f);
    int x0 = (int)sx; if (x0 > 62) x0 = 62;
    int y0 = (int)sy; if (y0 > 62) y0 = 62;
    float wx = sx - x0, wy = sy - y0;
    float v00 = img[y0 * 64 + x0],       v01 = img[y0 * 64 + x0 + 1];
    float v10 = img[(y0 + 1) * 64 + x0], v11 = img[(y0 + 1) * 64 + x0 + 1];
    return (v00 * (1 - wx) + v01 * wx) * (1 - wy) + (v10 * (1 - wx) + v11 * wx) * wy;
}
// value of the x4-upsampled image at integer pixel (ix,iy); clamped source taps
__device__ __forceinline__ float up_pixel(const float* img64, int ix, int iy) {
    return sample64_clamp(img64, (ix + 0.5f) * 0.25f - 0.5f, (iy + 0.5f) * 0.25f - 0.5f);
}
// point_sample (zero padded) of the *upsampled* 256x256 image, nested bilinear.
// Loads are unconditional on clamped indices; zero padding applied arithmetically
// (keeps EXEC uniform -> no saveexec/branch in the inner loops).
__device__ float sample_up(const float* img64, float cx, float cy) {
    float x = cx * 256.0f - 0.5f, y = cy * 256.0f - 0.5f;
    float xf = floorf(x), yf = floorf(y);
    int x0 = (int)xf, y0 = (int)yf;
    float wx = x - xf, wy = y - yf;
    float acc = 0.0f;
    #pragma unroll
    for (int dy = 0; dy < 2; ++dy) {
        #pragma unroll
        for (int dx = 0; dx < 2; ++dx) {
            int ix = x0 + dx, iy = y0 + dy;
            float inr = ((unsigned)ix < 256u && (unsigned)iy < 256u) ? 1.0f : 0.0f;
            int cix = min(max(ix, 0), 255), ciy = min(max(iy, 0), 255);
            float w = (dx ? wx : 1.0f - wx) * (dy ? wy : 1.0f - wy);
            acc += (w * inr) * up_pixel(img64, cix, ciy);
        }
    }
    return acc;
}
// point_sample (zero padded) of a stored 256x256 image; unconditional clamped loads
__device__ float sample256(const float* img, float cx, float cy) {
    float x = cx * 256.0f - 0.5f, y = cy * 256.0f - 0.5f;
    float xf = floorf(x), yf = floorf(y);
    int x0 = (int)xf, y0 = (int)yf;
    float wx = x - xf, wy = y - yf;
    float acc = 0.0f;
    #pragma unroll
    for (int dy = 0; dy < 2; ++dy) {
        #pragma unroll
        for (int dx = 0; dx < 2; ++dx) {
            int ix = x0 + dx, iy = y0 + dy;
            float inr = ((unsigned)ix < 256u && (unsigned)iy < 256u) ? 1.0f : 0.0f;
            int cix = min(max(ix, 0), 255), ciy = min(max(iy, 0), 255);
            float w = (dx ? wx : 1.0f - wx) * (dy ? wy : 1.0f - wy);
            acc += (w * inr) * img[ciy * 256 + cix];
        }
    }
    return acc;
}

// ---------------- generic zero ----------------
__global__ void zero_kernel(float* p, int n) {
    for (int i = blockIdx.x * blockDim.x + threadIdx.x; i < n; i += gridDim.x * blockDim.x)
        p[i] = 0.0f;
}

// ---------------- stage 1: matcher point sampling ----------------
__global__ void out_pts_kernel(const float* __restrict__ masks, float* __restrict__ out_pts, int li) {
    int idx = blockIdx.x * blockDim.x + threadIdx.x;
    if (idx >= Bc * Qc * Pc) return;
    int p = idx % Pc; int bq = idx / Pc; int b = bq / Qc;
    float cx = rng_u01(li * 4 + 0, (uint32_t)(b * Pc + p), 0);
    float cy = rng_u01(li * 4 + 0, (uint32_t)(b * Pc + p), 1);
    out_pts[idx] = sample_up(masks + (size_t)bq * (Hc * Hc), cx, cy);
}
__global__ void tgt_pts_kernel(const float* __restrict__ tgtm, float* __restrict__ tgt_pts, int li) {
    int idx = blockIdx.x * blockDim.x + threadIdx.x;
    if (idx >= Bc * Tc * Pc) return;
    int p = idx % Pc; int bt = idx / Pc; int b = bt / Tc;
    float cx = rng_u01(li * 4 + 0, (uint32_t)(b * Pc + p), 0);
    float cy = rng_u01(li * 4 + 0, (uint32_t)(b * Pc + p), 1);
    tgt_pts[idx] = sample256(tgtm + (size_t)bt * (UHc * UHc), cx, cy);
}
__global__ void tgt_sum_kernel(const float* __restrict__ tgt_pts, float* __restrict__ tgt_sum) {
    int bt = blockIdx.x;
    __shared__ float sh[256];
    float s = 0.0f;
    for (int p = threadIdx.x; p < Pc; p += 256) s += tgt_pts[(size_t)bt * Pc + p];
    sh[threadIdx.x] = s; __syncthreads();
    for (int off = 128; off > 0; off >>= 1) {
        if (threadIdx.x < off) sh[threadIdx.x] += sh[threadIdx.x + off];
        __syncthreads();
    }
    if (threadIdx.x == 0) tgt_sum[bt] = sh[0];
}

// ---------------- stage 2: cost GEMMs via WMMA ----------------
// One wave per (image, 16-query tile). Streams K=P in chunks of 32.
// acc0 = pos @ [tgt | ones@col10]   (col10 unused)
// acc1 = neg @ (1 - tgt)
// acc2 = sig @ [tgt | ones@col10]   (col10 = sig row sums, for dice denom)
// All lane loads are unconditional (rows clamped), padding handled by arithmetic
// masks so EXEC stays all-ones around the WMMAs (ISA requirement).
__global__ void __launch_bounds__(32) cost_wmma_kernel(const float* __restrict__ out_pts,
                                                       const float* __restrict__ tgt_pts,
                                                       float* __restrict__ cost_acc) {
    int b    = blockIdx.x / QT;
    int tile = blockIdx.x % QT;
    int lane = threadIdx.x;
    int m    = lane & 15;
    bool hi  = lane >= 16;
    int qrow = tile * 16 + m; if (qrow >= Qc) qrow = Qc - 1;  // padded rows ignored later
    const float* arow = out_pts + ((size_t)(b * Qc) + qrow) * Pc;
    const float* brow = tgt_pts + ((size_t)(b * Tc) + ((m < Tc) ? m : 0)) * Pc;
    const float bmask = (m < Tc) ? 1.0f : 0.0f;   // zero out padding columns arithmetically
    const float bone  = (m == Tc) ? 1.0f : 0.0f;  // ones column at t = 10

    v8f acc0 = {}; v8f acc1 = {}; v8f acc2 = {};
    for (int c = 0; c < Pc / 32; ++c) {
        int p0 = c * 32;
        v16h apos, aneg, asig, btg, b1m;
        #pragma unroll
        for (int e = 0; e < 16; ++e) {
            // A (16x32 f16) layout: lanes0-15 K = {0..7,16..23}, lanes16-31 K = {8..15,24..31}
            int ka = (e < 8) ? e : e + 8; if (hi) ka += 8;
            float x = arow[p0 + ka];
            apos[e] = (_Float16)softplusf(-x);
            aneg[e] = (_Float16)softplusf(x);
            asig[e] = (_Float16)sigmoidf(x);
            // B (32x16 f16) layout: lanes0-15 K = {0..7,16..23}, lanes16-31 K = {8..15,24..31} (lane = column N)
            int kb = ((e < 8) ? e : e + 8) + (hi ? 8 : 0);
            float tv = brow[p0 + kb];                       // unconditional load (row clamped)
            btg[e] = (_Float16)(bmask * tv + bone);
            b1m[e] = (_Float16)(bmask * (1.0f - tv));
        }
        acc0 = __builtin_amdgcn_wmma_f32_16x16x32_f16(false, apos, false, btg, (short)0, acc0, false, false);
        acc1 = __builtin_amdgcn_wmma_f32_16x16x32_f16(false, aneg, false, b1m, (short)0, acc1, false, false);
        acc2 = __builtin_amdgcn_wmma_f32_16x16x32_f16(false, asig, false, btg, (short)0, acc2, false, false);
    }
    float* dst = cost_acc + (size_t)(b * QT + tile) * 3 * 256;
    #pragma unroll
    for (int v = 0; v < 8; ++v) {
        dst[0 * 256 + v * 32 + lane] = acc0[v];
        dst[1 * 256 + v * 32 + lane] = acc1[v];
        dst[2 * 256 + v * 32 + lane] = acc2[v];
    }
}

// decode C/D fragment element (m,n): lane = n + (m>=8)*16, vgpr = m&7
__device__ __forceinline__ float cget(const float* cost_acc, int b, int q, int j, int n) {
    int tile = q >> 4; int mm = q & 15;
    int lane = n + ((mm >= 8) ? 16 : 0);
    return cost_acc[((size_t)(b * QT + tile) * 3 + j) * 256 + (mm & 7) * 32 + lane];
}

// ---------------- stage 3: Hungarian assignment (tiny, 1 thread/image) ----------------
__global__ void hungarian_kernel(const float* __restrict__ logits,
                                 const float* __restrict__ cost_acc,
                                 const float* __restrict__ tgt_sum,
                                 const int* __restrict__ tgt_labels,
                                 int* __restrict__ qidx) {
    int b = blockIdx.x;
    double C[Tc][Qc];
    for (int q = 0; q < Qc; ++q) {
        float l0 = logits[(size_t)(b * Qc + q) * 2 + 0];
        float l1 = logits[(size_t)(b * Qc + q) * 2 + 1];
        float mx = fmaxf(l0, l1);
        float e0 = expf(l0 - mx), e1 = expf(l1 - mx);
        float z = e0 + e1;
        float prob0 = e0 / z, prob1 = e1 / z;
        float sigRow = cget(cost_acc, b, q, 2, Tc);  // ones column -> sig.sum over P
        for (int t = 0; t < Tc; ++t) {
            float posT = cget(cost_acc, b, q, 0, t);
            float neg1 = cget(cost_acc, b, q, 1, t);
            float sigT = cget(cost_acc, b, q, 2, t);
            float cmask = (posT + neg1) / (float)Pc;
            float cdice = 1.0f - (2.0f * sigT + 1.0f) / (sigRow + tgt_sum[b * Tc + t] + 1.0f);
            int cls = tgt_labels[b * Tc + t];
            float cclass = -((cls == 0) ? prob0 : prob1);
            C[t][q] = 5.0 * (double)cmask + 2.0 * (double)cclass + 5.0 * (double)cdice;
        }
    }
    // Kuhn-Munkres with potentials; rows = T targets, cols = Q queries
    double u[Tc + 1], v[Qc + 1];
    int p[Qc + 1], way[Qc + 1];
    for (int i = 0; i <= Tc; ++i) u[i] = 0.0;
    for (int j = 0; j <= Qc; ++j) { v[j] = 0.0; p[j] = 0; way[j] = 0; }
    for (int i = 1; i <= Tc; ++i) {
        p[0] = i; int j0 = 0;
        double minv[Qc + 1]; bool used[Qc + 1];
        for (int j = 0; j <= Qc; ++j) { minv[j] = 1e18; used[j] = false; }
        while (true) {
            used[j0] = true; int i0 = p[j0]; int j1 = -1; double delta = 1e18;
            for (int j = 1; j <= Qc; ++j) if (!used[j]) {
                double cur = C[i0 - 1][j - 1] - u[i0] - v[j];
                if (cur < minv[j]) { minv[j] = cur; way[j] = j0; }
                if (minv[j] < delta) { delta = minv[j]; j1 = j; }
            }
            for (int j = 0; j <= Qc; ++j) {
                if (used[j]) { u[p[j]] += delta; v[j] -= delta; }
                else minv[j] -= delta;
            }
            j0 = j1;
            if (p[j0] == 0) break;
        }
        while (j0) { int j1 = way[j0]; p[j0] = p[j1]; j0 = j1; }
    }
    for (int j = 1; j <= Qc; ++j) if (p[j] > 0) qidx[b * Tc + (p[j] - 1)] = j - 1;
}

// ---------------- stage 4: weighted CE loss ----------------
__global__ void ce_kernel(const float* __restrict__ logits,
                          const int* __restrict__ tgt_labels,
                          const int* __restrict__ qidx,
                          float* __restrict__ out) {
    __shared__ float s_wce[256], s_w[256];
    float wce = 0.0f, wsum = 0.0f;
    for (int e = threadIdx.x; e < Bc * Qc; e += 256) {
        int b = e / Qc, q = e % Qc;
        int tc = 1;  // no-object class
        for (int t = 0; t < Tc; ++t)
            if (qidx[b * Tc + t] == q) tc = tgt_labels[b * Tc + t];
        float l0 = logits[(size_t)e * 2 + 0], l1 = logits[(size_t)e * 2 + 1];
        float mx = fmaxf(l0, l1);
        float lse = mx + logf(expf(l0 - mx) + expf(l1 - mx));
        float lt = (tc == 0) ? l0 : l1;
        float w = (tc == 1) ? 0.1f : 1.0f;
        wce += w * (lse - lt);
        wsum += w;
    }
    s_wce[threadIdx.x] = wce; s_w[threadIdx.x] = wsum; __syncthreads();
    for (int off = 128; off > 0; off >>= 1) {
        if (threadIdx.x < off) { s_wce[threadIdx.x] += s_wce[threadIdx.x + off]; s_w[threadIdx.x] += s_w[threadIdx.x + off]; }
        __syncthreads();
    }
    if (threadIdx.x == 0) atomicAdd(out, 2.0f * (s_wce[0] / s_w[0]));  // CLASS_W * loss_ce
}

// ---------------- stage 5: uncertainty sampling ----------------
__global__ void unc_kernel(const float* __restrict__ masks,
                           const int* __restrict__ qidx,
                           float* __restrict__ unc, int li) {
    int idx = blockIdx.x * blockDim.x + threadIdx.x;
    if (idx >= Nc * NSc) return;
    int n = idx / NSc;
    int b = n / Tc;
    int q = qidx[n];
    float cx = rng_u01(li * 4 + 1, (uint32_t)idx, 0);
    float cy = rng_u01(li * 4 + 1, (uint32_t)idx, 1);
    float v = sample_up(masks + (size_t)(b * Qc + q) * (Hc * Hc), cx, cy);
    unc[idx] = -fabsf(v);
}

// per-mask top-KIMP selection via threshold bisection + compaction
__global__ void select_kernel(const float* __restrict__ unc, int* __restrict__ sel) {
    int n = blockIdx.x;
    const float* u = unc + (size_t)n * NSc;
    __shared__ int red[256];
    __shared__ int c1, c2;
    float lo = -64.0f, hi = 0.0f;
    for (int it = 0; it < 28; ++it) {
        float mid = 0.5f * (lo + hi);
        int c = 0;
        for (int s = threadIdx.x; s < NSc; s += 256) c += (u[s] > mid) ? 1 : 0;
        red[threadIdx.x] = c; __syncthreads();
        for (int off = 128; off > 0; off >>= 1) {
            if (threadIdx.x < off) red[threadIdx.x] += red[threadIdx.x + off];
            __syncthreads();
        }
        int total = red[0]; __syncthreads();
        if (total > KIMP) lo = mid; else hi = mid;
    }
    if (threadIdx.x == 0) { c1 = 0; c2 = 0; } __syncthreads();
    for (int s = threadIdx.x; s < NSc; s += 256)
        if (u[s] > hi) {
            int pos = atomicAdd(&c1, 1);
            if (pos < KIMP) sel[(size_t)n * KIMP + pos] = s;
        }
    __syncthreads();
    int base = (c1 < KIMP) ? c1 : KIMP;
    __syncthreads();
    for (int s = threadIdx.x; s < NSc; s += 256)
        if (u[s] <= hi && u[s] > lo) {
            int pos = atomicAdd(&c2, 1);
            if (base + pos < KIMP) sel[(size_t)n * KIMP + base + pos] = s;
        }
    __syncthreads();
    if (threadIdx.x == 0) {
        int f2 = c2; if (f2 > KIMP - base) f2 = KIMP - base;
        for (int i = base + f2; i < KIMP; ++i) sel[(size_t)n * KIMP + i] = i;  // safety pad
    }
}

// ---------------- stage 6: BCE + dice accumulation ----------------
__global__ void maskloss_kernel(const float* __restrict__ masks,
                                const float* __restrict__ tgtm,
                                const int* __restrict__ qidx,
                                const int* __restrict__ sel,
                                float* __restrict__ accn, int li) {
    int n = blockIdx.y;
    int p = blockIdx.x * 256 + threadIdx.x;  // P = 49*256 exactly
    int b = n / Tc, t = n % Tc;
    int q = qidx[n];
    float cx, cy;
    if (p < KIMP) {
        int s = sel[(size_t)n * KIMP + p];
        cx = rng_u01(li * 4 + 1, (uint32_t)(n * NSc + s), 0);
        cy = rng_u01(li * 4 + 1, (uint32_t)(n * NSc + s), 1);
    } else {
        int r = p - KIMP;
        cx = rng_u01(li * 4 + 2, (uint32_t)(n * KRANDc + r), 0);
        cy = rng_u01(li * 4 + 2, (uint32_t)(n * KRANDc + r), 1);
    }
    float plog = sample_up(masks + (size_t)(b * Qc + q) * (Hc * Hc), cx, cy);
    float lab  = sample256(tgtm + (size_t)(b * Tc + t) * (UHc * UHc), cx, cy);
    float bce = softplusf(plog) - plog * lab;
    float sg = sigmoidf(plog);
    __shared__ float sh[4][256];
    sh[0][threadIdx.x] = bce;
    sh[1][threadIdx.x] = sg;
    sh[2][threadIdx.x] = sg * lab;
    sh[3][threadIdx.x] = lab;
    __syncthreads();
    for (int off = 128; off > 0; off >>= 1) {
        if (threadIdx.x < off)
            #pragma unroll
            for (int k = 0; k < 4; ++k) sh[k][threadIdx.x] += sh[k][threadIdx.x + off];
        __syncthreads();
    }
    if (threadIdx.x == 0)
        #pragma unroll
        for (int k = 0; k < 4; ++k) atomicAdd(&accn[n * 4 + k], sh[k][0]);
}

__global__ void combine_kernel(const float* __restrict__ accn, float* __restrict__ out) {
    __shared__ float sm[32], sd[32];
    int n = threadIdx.x;
    float lm = 0.0f, ld = 0.0f;
    if (n < Nc) {
        float bce = accn[n * 4 + 0], sg = accn[n * 4 + 1];
        float sl = accn[n * 4 + 2], lab = accn[n * 4 + 3];
        lm = bce / (float)Pc;
        ld = 1.0f - (2.0f * sl + 1.0f) / (sg + lab + 1.0f);
    }
    sm[n] = lm; sd[n] = ld; __syncthreads();
    for (int off = 16; off > 0; off >>= 1) {
        if (n < off) { sm[n] += sm[n + off]; sd[n] += sd[n + off]; }
        __syncthreads();
    }
    if (n == 0)
        atomicAdd(out, 5.0f * (sm[0] / (float)Nc) + 5.0f * (sd[0] / (float)Nc));  // MASK_W, DICE_W
}

// ---------------- host launch ----------------
extern "C" void kernel_launch(void* const* d_in, const int* in_sizes, int n_in,
                              void* d_out, int out_size, void* d_ws, size_t ws_size,
                              hipStream_t stream) {
    const float* pred_logits = (const float*)d_in[0];
    const float* pred_masks  = (const float*)d_in[1];
    const float* aux_logits  = (const float*)d_in[2];
    const float* aux_masks   = (const float*)d_in[3];
    const int*   tgt_labels  = (const int*)d_in[4];
    const float* tgt_masks   = (const float*)d_in[5];
    float* out = (float*)d_out;
    float* ws  = (float*)d_ws;

    // workspace layout (float slots), total ~3.71M floats ≈ 14.8 MB
    size_t o_out_pts = 0;
    size_t o_tgt_pts = o_out_pts + (size_t)Bc * Qc * Pc;
    size_t o_tgt_sum = o_tgt_pts + (size_t)Bc * Tc * Pc;
    size_t o_cost    = o_tgt_sum + (size_t)Bc * Tc;
    size_t o_qidx    = o_cost    + (size_t)Bc * QT * 3 * 256;
    size_t o_unc     = o_qidx    + (size_t)Bc * Tc;
    size_t o_sel     = o_unc     + (size_t)Nc * NSc;
    size_t o_accn    = o_sel     + (size_t)Nc * KIMP;

    float* out_pts  = ws + o_out_pts;
    float* tgt_pts  = ws + o_tgt_pts;
    float* tgt_sum  = ws + o_tgt_sum;
    float* cost_acc = ws + o_cost;
    int*   qidx     = (int*)(ws + o_qidx);
    float* unc      = ws + o_unc;
    int*   sel      = (int*)(ws + o_sel);
    float* accn     = ws + o_accn;

    zero_kernel<<<1, 32, 0, stream>>>(out, out_size);

    for (int li = 0; li < LAYERS; ++li) {
        const float* lg = (li == 0) ? pred_logits : aux_logits + (size_t)(li - 1) * Bc * Qc * 2;
        const float* mk = (li == 0) ? pred_masks  : aux_masks  + (size_t)(li - 1) * Bc * Qc * Hc * Hc;

        out_pts_kernel<<<(Bc * Qc * Pc + 255) / 256, 256, 0, stream>>>(mk, out_pts, li);
        tgt_pts_kernel<<<(Bc * Tc * Pc + 255) / 256, 256, 0, stream>>>(tgt_masks, tgt_pts, li);
        tgt_sum_kernel<<<Bc * Tc, 256, 0, stream>>>(tgt_pts, tgt_sum);
        cost_wmma_kernel<<<Bc * QT, 32, 0, stream>>>(out_pts, tgt_pts, cost_acc);
        hungarian_kernel<<<Bc, 1, 0, stream>>>(lg, cost_acc, tgt_sum, tgt_labels, qidx);
        ce_kernel<<<1, 256, 0, stream>>>(lg, tgt_labels, qidx, out);
        unc_kernel<<<(Nc * NSc + 255) / 256, 256, 0, stream>>>(mk, qidx, unc, li);
        select_kernel<<<Nc, 256, 0, stream>>>(unc, sel);
        zero_kernel<<<1, 128, 0, stream>>>(accn, Nc * 4);
        maskloss_kernel<<<dim3(Pc / 256, Nc), 256, 0, stream>>>(mk, tgt_masks, qidx, sel, accn, li);
        combine_kernel<<<1, 32, 0, stream>>>(accn, out);
    }
}
